// MultiheadAttention_21569325761245
// MI455X (gfx1250) — matrix operand order repaired and checked by reading, not compiled, not run on hardware
//
#include <hip/hip_runtime.h>
#include <hip/hip_bf16.h>
#include <math.h>

typedef __attribute__((ext_vector_type(16))) __bf16 v16bf;
typedef __attribute__((ext_vector_type(8)))  __bf16 v8bf;
typedef __attribute__((ext_vector_type(4)))  __bf16 v4bf;
typedef __attribute__((ext_vector_type(8)))  float  v8f;
typedef __attribute__((ext_vector_type(4)))  float  v4f;

union Frag16 { v16bf v; v8bf h[2]; };

#define BB   2
#define SS   2048
#define DD   2048
#define HH   16
#define HD   128
#define ND3  6144
#define MTOT 4096   // B*S

__device__ __forceinline__ v8f wmma_bf16(v16bf a, v16bf b, v8f c) {
  // D = A(16x32 bf16) * B(32x16 bf16) + C(16x16 f32)
  return __builtin_amdgcn_wmma_f32_16x16x32_bf16(false, a, false, b, (short)0, c, false, false);
}

// ---------------------------------------------------------------------------
// f32 -> bf16 conversion (vectorized 4-wide)
// ---------------------------------------------------------------------------
__global__ void cvt_f32_bf16(const float* __restrict__ in, __bf16* __restrict__ out, int n4) {
  int i = blockIdx.x * blockDim.x + threadIdx.x;
  if (i >= n4) return;
  v4f x = ((const v4f*)in)[i];
  v4bf y;
  y[0] = (__bf16)x[0];
  y[1] = (__bf16)x[1];
  y[2] = (__bf16)x[2];
  y[3] = (__bf16)x[3];
  ((v4bf*)out)[i] = y;
}

// ---------------------------------------------------------------------------
// QKV GEMM: qkv[m,n] = sum_k x[m,k]*Wqkv[n,k] + b[n]
// Wave = 16(M) x 64(N); block = 8 waves -> 128(M) x 64(N)
// Epilogue scatters into qkv[t][b][h][s][hd] bf16
// ---------------------------------------------------------------------------
__global__ void __launch_bounds__(256)
gemm_qkv_kernel(const __bf16* __restrict__ A,
                const __bf16* __restrict__ W,
                const float*  __restrict__ bias,
                __bf16* __restrict__ qkv)
{
  const int lane = threadIdx.x & 31;
  const int wave = threadIdx.x >> 5;
  const int m0 = blockIdx.y * 128 + wave * 16;
  const int n0 = blockIdx.x * 64;

  v8f acc[4] = {};

  const __bf16* Arow = A + (size_t)(m0 + (lane & 15)) * DD;
  const int akb  = (lane >> 4) << 4;   // A frag K sub-offset per lane half
  const int bcol = lane & 15;          // B frag column
  const int bkb  = (lane >> 4) << 4;   // B frag K sub-offset per lane half

  for (int k = 0; k < DD; k += 32) {
    Frag16 a;
    a.h[0] = *(const v8bf*)(Arow + k + akb);
    a.h[1] = *(const v8bf*)(Arow + k + akb + 8);
#pragma unroll
    for (int c = 0; c < 4; ++c) {
      Frag16 bf;
      bf.v = *(const v16bf*)(W + (size_t)(n0 + c * 16 + bcol) * DD + k + bkb);
      acc[c] = wmma_bf16(a.v, bf.v, acc[c]);
    }
  }

#pragma unroll
  for (int c = 0; c < 4; ++c) {
    const int n   = n0 + c * 16 + bcol;
    const int t   = n >> 11;        // 0=Q, 1=K, 2=V
    const int rem = n & 2047;
    const int h   = rem >> 7;
    const int hd  = rem & 127;
    const float bb = bias[n];
#pragma unroll
    for (int r = 0; r < 8; ++r) {
      const int m = m0 + r + ((lane >> 4) << 3);  // C layout: lanes>=16 hold M+8
      const int b = m >> 11;
      const int s = m & 2047;
      qkv[(((((size_t)t * BB + b) * HH) + h) * SS + s) * HD + hd] = (__bf16)(acc[c][r] + bb);
    }
  }
}

// ---------------------------------------------------------------------------
// V transpose: [B*H][S][HD] -> [B*H][HD][S], LDS-tiled, coalesced both sides
// ---------------------------------------------------------------------------
__global__ void __launch_bounds__(256)
transpose_v_kernel(const __bf16* __restrict__ V,  // [B*H][S][HD]
                   __bf16* __restrict__ vT)       // [B*H][HD][S]
{
  __shared__ __bf16 T[64][72];                    // +8 pad vs bank conflicts
  const int plane = blockIdx.z;                   // b*H + h
  const int s0  = blockIdx.x * 64;
  const int hd0 = blockIdx.y * 64;
  const int t = threadIdx.x;

  {
    const int r    = t >> 2;          // 0..63 : s row
    const int cseg = (t & 3) * 16;    // hd segment
    const __bf16* src = V + ((size_t)plane * SS + s0 + r) * HD + hd0 + cseg;
    v8bf a0 = *(const v8bf*)(src);
    v8bf a1 = *(const v8bf*)(src + 8);
#pragma unroll
    for (int j = 0; j < 8; ++j) { T[r][cseg + j] = a0[j]; T[r][cseg + 8 + j] = a1[j]; }
  }
  __syncthreads();
  {
    const int c    = t >> 2;          // 0..63 : hd row (output)
    const int rseg = (t & 3) * 16;    // s segment
    v8bf o0, o1;
#pragma unroll
    for (int j = 0; j < 8; ++j) { o0[j] = T[rseg + j][c]; o1[j] = T[rseg + 8 + j][c]; }
    __bf16* dst = vT + ((size_t)plane * HD + hd0 + c) * SS + s0 + rseg;
    *(v8bf*)dst = o0;
    *(v8bf*)(dst + 8) = o1;
  }
}

// ---------------------------------------------------------------------------
// Flash attention, causal, additive bias.
// Block = 4 independent waves; each wave owns a 16-row q tile and iterates
// only to its own causal frontier. No barriers: P tile LDS is per-wave
// (same-wave DS ordering), V^T comes pre-transposed from global.
// ---------------------------------------------------------------------------
__global__ void __launch_bounds__(128)
attn_kernel(const __bf16* __restrict__ qkv,
            const __bf16* __restrict__ vT,     // [B,H,HD,S]
            const float*  __restrict__ bias,   // [1,H,S,S]
            __bf16* __restrict__ outb)         // [B,S,D] with D = h*128+hd
{
  __shared__ __bf16 Plds[4 * 16 * 32];         // per-wave P tile (16q x 32k)

  const int lane = threadIdx.x & 31;
  const int wave = threadIdx.x >> 5;
  const int h = blockIdx.y;
  const int b = blockIdx.z;
  const int q0 = blockIdx.x * 64 + wave * 16;

  const size_t plane = (size_t)(b * HH + h) * SS * HD;
  const __bf16* Q  = qkv + plane;
  const __bf16* K  = qkv + (size_t)BB * HH * SS * HD + plane;
  const __bf16* Vt = vT + plane;               // same plane size, [HD][S]

  // Q fragments for the whole 16x128 tile (4 K-chunks of 32)
  Frag16 qf[4];
  {
    const __bf16* qrow = Q + (size_t)(q0 + (lane & 15)) * HD;
    const int akb = (lane >> 4) << 4;
#pragma unroll
    for (int c = 0; c < 4; ++c) {
      qf[c].h[0] = *(const v8bf*)(qrow + c * 32 + akb);
      qf[c].h[1] = *(const v8bf*)(qrow + c * 32 + akb + 8);
    }
  }

  v8f o[8] = {};
  float m_i[8], l_i[8];
#pragma unroll
  for (int r = 0; r < 8; ++r) { m_i[r] = -3.0e38f; l_i[r] = 0.f; }

  const float scl = 0.08838834764831845f; // 1/sqrt(128)
  const int nkt = (q0 >> 5) + 1;          // exact causal frontier of this wave
  __bf16* Pw = Plds + wave * (16 * 32);

  for (int kt = 0; kt < nkt; ++kt) {
    const int k0 = kt * 32;

    // Scores: two 16-key tiles, each 4 chained WMMAs over hd=128
    v8f sc[2];
#pragma unroll
    for (int t = 0; t < 2; ++t) {
      v8f s = {};
      const __bf16* krow = K + (size_t)(k0 + t * 16 + (lane & 15)) * HD;
      const int bkb = (lane >> 4) << 4;
#pragma unroll
      for (int c = 0; c < 4; ++c) {
        Frag16 kf;
        kf.v = *(const v16bf*)(krow + c * 32 + bkb);
        s = wmma_bf16(qf[c].v, kf.v, s);
      }
      const int key = k0 + t * 16 + (lane & 15);
#pragma unroll
      for (int r = 0; r < 8; ++r) {
        const int row = q0 + r + ((lane >> 4) << 3);
        float v = s[r] * scl + bias[((size_t)h * SS + row) * SS + key];
        s[r] = (key <= row) ? v : -3.0e38f;
      }
      sc[t] = s;
    }

    // Online softmax: row groups are 16 lanes; reduce with xor masks 1..8
    float alpha[8];
#pragma unroll
    for (int r = 0; r < 8; ++r) {
      float mx = fmaxf(sc[0][r], sc[1][r]);
      mx = fmaxf(mx, __shfl_xor(mx, 1));
      mx = fmaxf(mx, __shfl_xor(mx, 2));
      mx = fmaxf(mx, __shfl_xor(mx, 4));
      mx = fmaxf(mx, __shfl_xor(mx, 8));
      const float mnew = fmaxf(m_i[r], mx);
      const float al = __expf(m_i[r] - mnew);
      const float p0 = __expf(sc[0][r] - mnew);
      const float p1 = __expf(sc[1][r] - mnew);
      float rs = p0 + p1;
      rs += __shfl_xor(rs, 1);
      rs += __shfl_xor(rs, 2);
      rs += __shfl_xor(rs, 4);
      rs += __shfl_xor(rs, 8);
      l_i[r] = l_i[r] * al + rs;
      m_i[r] = mnew;
      alpha[r] = al;
      sc[0][r] = p0;
      sc[1][r] = p1;
    }
#pragma unroll
    for (int c = 0; c < 8; ++c)
#pragma unroll
      for (int r = 0; r < 8; ++r) o[c][r] *= alpha[r];

    // Stage P (bf16) row-major [16q x 32k] for A-fragment reload (per-wave LDS)
#pragma unroll
    for (int t = 0; t < 2; ++t)
#pragma unroll
      for (int r = 0; r < 8; ++r) {
        const int row = r + ((lane >> 4) << 3);
        Pw[row * 32 + t * 16 + (lane & 15)] = (__bf16)sc[t][r];
      }

    Frag16 pa;
    {
      const int prow = lane & 15;
      const int akb = (lane >> 4) << 4;
      pa.h[0] = *(const v8bf*)(Pw + prow * 32 + akb);
      pa.h[1] = *(const v8bf*)(Pw + prow * 32 + akb + 8);
    }

    // O += P (16x32) @ V (32x128): 8 WMMAs, B-fragments straight from V^T
#pragma unroll
    for (int c = 0; c < 8; ++c) {
      const int hd = c * 16 + (lane & 15);
      const int kb = k0 + ((lane >> 4) << 4);
      Frag16 vf;
      vf.v = *(const v16bf*)(Vt + (size_t)hd * SS + kb);
      o[c] = wmma_bf16(pa.v, vf.v, o[c]);
    }
  }

  // Normalize and store to [b, s, h*128+hd] bf16
#pragma unroll
  for (int r = 0; r < 8; ++r) l_i[r] = 1.f / l_i[r];
#pragma unroll
  for (int c = 0; c < 8; ++c) {
    const int hd = c * 16 + (lane & 15);
#pragma unroll
    for (int r = 0; r < 8; ++r) {
      const int s = q0 + r + ((lane >> 4) << 3);
      outb[((size_t)b * SS + s) * DD + h * HD + hd] = (__bf16)(o[c][r] * l_i[r]);
    }
  }
}

// ---------------------------------------------------------------------------
// Output projection: out[m,n] = sum_k attn[m,k]*Wout[n,k] + bout[n]  (f32 out)
// ---------------------------------------------------------------------------
__global__ void __launch_bounds__(256)
gemm_out_kernel(const __bf16* __restrict__ A,   // [4096,2048] bf16
                const __bf16* __restrict__ W,   // [2048,2048] bf16
                const float*  __restrict__ bias,
                float* __restrict__ out)        // [4096,2048] f32
{
  const int lane = threadIdx.x & 31;
  const int wave = threadIdx.x >> 5;
  const int m0 = blockIdx.y * 128 + wave * 16;
  const int n0 = blockIdx.x * 64;

  v8f acc[4] = {};
  const __bf16* Arow = A + (size_t)(m0 + (lane & 15)) * DD;
  const int akb  = (lane >> 4) << 4;
  const int bcol = lane & 15;
  const int bkb  = (lane >> 4) << 4;

  for (int k = 0; k < DD; k += 32) {
    Frag16 a;
    a.h[0] = *(const v8bf*)(Arow + k + akb);
    a.h[1] = *(const v8bf*)(Arow + k + akb + 8);
#pragma unroll
    for (int c = 0; c < 4; ++c) {
      Frag16 bf;
      bf.v = *(const v16bf*)(W + (size_t)(n0 + c * 16 + bcol) * DD + k + bkb);
      acc[c] = wmma_bf16(a.v, bf.v, acc[c]);
    }
  }

#pragma unroll
  for (int c = 0; c < 4; ++c) {
    const int n = n0 + c * 16 + bcol;
    const float bb = bias[n];
#pragma unroll
    for (int r = 0; r < 8; ++r) {
      const int m = m0 + r + ((lane >> 4) << 3);
      out[(size_t)m * DD + n] = acc[c][r] + bb;
    }
  }
}

// ---------------------------------------------------------------------------
// Launch
// ---------------------------------------------------------------------------
extern "C" void kernel_launch(void* const* d_in, const int* in_sizes, int n_in,
                              void* d_out, int out_size, void* d_ws, size_t ws_size,
                              hipStream_t stream) {
  const float* x    = (const float*)d_in[0];
  const float* ab   = (const float*)d_in[1];
  const float* wqkv = (const float*)d_in[2];
  const float* bqkv = (const float*)d_in[3];
  const float* wout = (const float*)d_in[4];
  const float* bout = (const float*)d_in[5];
  float* out = (float*)d_out;

  char* ws = (char*)d_ws;
  __bf16* xb    = (__bf16*)(ws);                 // 16,777,216 B
  __bf16* wqkvb = (__bf16*)(ws + 16777216);      // 25,165,824 B
  __bf16* woutb = (__bf16*)(ws + 41943040);      //  8,388,608 B
  __bf16* qkvb  = (__bf16*)(ws + 50331648);      // 50,331,648 B
  __bf16* attnb = (__bf16*)(ws + 100663296);     // 16,777,216 B
  __bf16* vTb   = (__bf16*)(ws + 117440512);     // 16,777,216 B (total 128 MB)

  int n4 = MTOT * DD / 4;
  cvt_f32_bf16<<<(n4 + 255) / 256, 256, 0, stream>>>(x, xb, n4);
  n4 = ND3 * DD / 4;
  cvt_f32_bf16<<<(n4 + 255) / 256, 256, 0, stream>>>(wqkv, wqkvb, n4);
  n4 = DD * DD / 4;
  cvt_f32_bf16<<<(n4 + 255) / 256, 256, 0, stream>>>(wout, woutb, n4);

  gemm_qkv_kernel<<<dim3(ND3 / 64, MTOT / 128), 256, 0, stream>>>(xb, wqkvb, bqkv, qkvb);

  const __bf16* vsrc = qkvb + 2 * (size_t)BB * HH * SS * HD;
  transpose_v_kernel<<<dim3(SS / 64, HD / 64, BB * HH), 256, 0, stream>>>(vsrc, vTb);

  attn_kernel<<<dim3(SS / 64, HH, BB), 128, 0, stream>>>(qkvb, vTb, ab, attnb);
  gemm_out_kernel<<<dim3(DD / 64, MTOT / 128), 256, 0, stream>>>(attnb, woutb, bout, out);
}
